// HyperNet_89816356094515
// MI455X (gfx1250) — matrix operand order repaired and checked by reading, not compiled, not run on hardware
//
#include <hip/hip_runtime.h>
#include <string.h>

// ---------------------------------------------------------------------------
// Types for WMMA fragments / TDM descriptors (gfx1250, wave32)
// ---------------------------------------------------------------------------
typedef __attribute__((ext_vector_type(16))) __bf16 v16bf;
typedef __attribute__((ext_vector_type(8)))  __bf16 v8bf;
typedef __attribute__((ext_vector_type(4)))  __bf16 v4bf;
typedef __attribute__((ext_vector_type(8)))  float  v8f;
typedef __attribute__((ext_vector_type(4)))  unsigned int v4u;
typedef __attribute__((ext_vector_type(8)))  int    v8i;
typedef __attribute__((ext_vector_type(4)))  int    v4i;

#define HID    1024
#define OUTD   2304
#define MTOT   12160
#define NLAYER 13

__device__ __host__ inline __bf16 f32_to_bf16_bits(float f) {
    unsigned u;
    __builtin_memcpy(&u, &f, 4);
    unsigned r = (u + 0x7FFFu + ((u >> 16) & 1u)) >> 16;   // round-to-nearest-even
    unsigned short h = (unsigned short)r;
    __bf16 out;
    __builtin_memcpy(&out, &h, 2);
    return out;
}

// ---------------------------------------------------------------------------
// Tensor Data Mover: 2D tile load (bf16 elements) into LDS.
// D# layout per CDNA5 ISA ch.8: group0 = {flags, lds_addr, global_addr, type},
// group1 = {mask/data_size, tensor dims, tile dims, strides}. Groups 2/3 zero
// (2D tensor). data_size=1 -> 2-byte elements.
// ---------------------------------------------------------------------------
__device__ inline void tdm_load_2d_bf16(unsigned lds_off_bytes, const void* gptr,
                                        unsigned tensor_d0, unsigned tensor_d1,
                                        unsigned tile_d0, unsigned tile_d1,
                                        unsigned long long d0_stride)
{
    unsigned long long ga = (unsigned long long)(uintptr_t)gptr;
    v4u g0;
    g0.x = 1u;                                        // count=1 (valid), user mode
    g0.y = lds_off_bytes;                             // lds_addr (bytes)
    g0.z = (unsigned)(ga & 0xFFFFFFFFu);              // global_addr[31:0]
    g0.w = (unsigned)((ga >> 32) & 0x01FFFFFFu)       // global_addr[56:32]
         | (2u << 30);                                // type=2 ("image")
    v8i g1;
    g1[0] = (int)(1u << 16);                          // wg_mask=0, data_size=1 (2B)
    g1[1] = (int)(tensor_d0 << 16);                   // barrier_addr=0 | dim0 lo16
    g1[2] = (int)((tensor_d0 >> 16) | (tensor_d1 << 16)); // dim0 hi16 | dim1 lo16
    g1[3] = (int)((tensor_d1 >> 16) | (tile_d0 << 16));   // dim1 hi16 | tile_dim0
    g1[4] = (int)(tile_d1 & 0xFFFFu);                 // tile_dim1 | tile_dim2=0
    g1[5] = (int)(unsigned)(d0_stride & 0xFFFFFFFFull);   // dim0_stride lo32
    g1[6] = (int)(unsigned)((d0_stride >> 32) & 0xFFFFull); // dim0_stride hi16
    g1[7] = 0;
    v4i gz = {0, 0, 0, 0};
#if __clang_major__ >= 23
    v8i gz8 = {0, 0, 0, 0, 0, 0, 0, 0};
    __builtin_amdgcn_tensor_load_to_lds(g0, g1, gz, gz, gz8, 0);
#else
    __builtin_amdgcn_tensor_load_to_lds(g0, g1, gz, gz, 0);
#endif
}

// ---------------------------------------------------------------------------
// Prep kernels
// ---------------------------------------------------------------------------

// hbias[j] = b1[j] + sum_t input_emb[t] * W1[t][j]   (shared across all pairs)
__global__ __launch_bounds__(256) void hbias_kernel(
    const float* __restrict__ ie, const float* __restrict__ W1,
    const float* __restrict__ b1, float* __restrict__ hbias)
{
    int j = blockIdx.x * 256 + threadIdx.x;
    if (j >= HID) return;
    float s = b1[j];
    for (int t = 0; t < 512; ++t)
        s = fmaf(ie[t], W1[t * HID + j], s);
    hbias[j] = s;
}

// W1T[n][k] = bf16(W1[512+k][n]),  n<1024, k<64  (B operand: K-contiguous)
__global__ __launch_bounds__(256) void w1t_kernel(
    const float* __restrict__ W1, __bf16* __restrict__ W1T)
{
    int idx = blockIdx.x * 256 + threadIdx.x;
    if (idx >= HID * 64) return;
    int n = idx >> 6, k = idx & 63;
    W1T[idx] = f32_to_bf16_bits(W1[(512 + k) * HID + n]);
}

// W2T[n][k] = bf16(W2[k][n]),  n<2304, k<1024
__global__ __launch_bounds__(256) void w2t_kernel(
    const float* __restrict__ W2, __bf16* __restrict__ W2T)
{
    int idx = blockIdx.x * 256 + threadIdx.x;
    if (idx >= OUTD * HID) return;
    int n = idx >> 10, k = idx & 1023;
    W2T[idx] = f32_to_bf16_bits(W2[k * OUTD + n]);
}

// generic f32 -> bf16 cast
__global__ __launch_bounds__(256) void cvt_kernel(
    const float* __restrict__ src, __bf16* __restrict__ dst, int count)
{
    int i = blockIdx.x * 256 + threadIdx.x;
    if (i < count) dst[i] = f32_to_bf16_bits(src[i]);
}

// Per block-pair row m: output base offset and CI*9 stride for the
// transpose(0,2,1,3,4,5) scatter.
__global__ __launch_bounds__(256) void meta_kernel(
    int* __restrict__ row_base, int* __restrict__ ci9)
{
    int m = blockIdx.x * 256 + threadIdx.x;
    if (m >= MTOT) return;
    const int lstart[NLAYER + 1] = {0, 64, 192, 448, 960, 1984, 4032, 8128,
                                    10176, 11200, 11712, 11968, 12096, 12160};
    const int lbi[NLAYER] = {8, 8, 16, 16, 32, 32, 64, 64, 32, 32, 16, 16, 8};
    int L = 0;
    while (m >= lstart[L + 1]) ++L;
    int ml = m - lstart[L];
    int BI = lbi[L];
    int bo = ml / BI;
    int bi = ml - bo * BI;
    int CI = BI * 16;
    row_base[m] = lstart[L] * OUTD + (bo * 16 * CI + bi * 16) * 9;
    ci9[m]      = CI * 9;
}

// ---------------------------------------------------------------------------
// GEMM1: H[12160,1024] = relu(P[12160,64] @ W1p[64,1024] + hbias)  (bf16 out)
// Small (K=64): direct global-load fragments, 2 K-chunks fully unrolled.
// ---------------------------------------------------------------------------
__global__ __launch_bounds__(256) void gemm1_kernel(
    const __bf16* __restrict__ P, const __bf16* __restrict__ W1T,
    const float* __restrict__ hbias, __bf16* __restrict__ H)
{
    __shared__ float smem[8 * 16 * 128];
    const int lane = threadIdx.x & 31;
    const int wave = threadIdx.x >> 5;
    const int hi   = lane >> 4;
    const int ln16 = lane & 15;
    const int n0   = blockIdx.x * 128;
    const int m0   = (blockIdx.y * 8 + wave) * 16;

    v8f acc[8] = {};
    const __bf16* arow  = P   + (size_t)(m0 + ln16) * 64 + hi * 16;
    const __bf16* bbase = W1T + (size_t)(n0 + ln16) * 64 + hi * 16;

#pragma unroll
    for (int kc = 0; kc < 2; ++kc) {
        v8bf alo = *(const v8bf*)(arow + kc * 32);
        v8bf ahi = *(const v8bf*)(arow + kc * 32 + 8);
        v16bf a = __builtin_shufflevector(alo, ahi,
            0,1,2,3,4,5,6,7,8,9,10,11,12,13,14,15);
#pragma unroll
        for (int nt = 0; nt < 8; ++nt) {
            v16bf b = *(const v16bf*)(bbase + (size_t)nt * 16 * 64 + kc * 32);
            acc[nt] = __builtin_amdgcn_wmma_f32_16x16x32_bf16(
                false, a, false, b, (short)0, acc[nt], false, false);
        }
    }

    float* sw = smem + wave * (16 * 128);
#pragma unroll
    for (int nt = 0; nt < 8; ++nt)
#pragma unroll
        for (int r = 0; r < 8; ++r)
            sw[(hi * 8 + r) * 128 + nt * 16 + ln16] = acc[nt][r];
    __syncthreads();

    float hb[4];
#pragma unroll
    for (int c = 0; c < 4; ++c) hb[c] = hbias[n0 + lane * 4 + c];

    for (int row = 0; row < 16; ++row) {
        int m = m0 + row;
        v4bf hv;
#pragma unroll
        for (int c = 0; c < 4; ++c) {
            float v = sw[row * 128 + lane * 4 + c] + hb[c];
            v = fmaxf(v, 0.0f);
            hv[c] = f32_to_bf16_bits(v);
        }
        *(v4bf*)(H + (size_t)m * HID + n0 + lane * 4) = hv;
    }
}

// ---------------------------------------------------------------------------
// GEMM2: OUT = scatter(H[12160,1024] @ W2[1024,2304] + b2)
// TDM double-buffered pipeline: wave 0 DMAs the 128x32 A (H) and B (W2T)
// tiles for chunk kc+1 into LDS while all 8 waves run WMMA on chunk kc.
// Epilogue stages through LDS -> contiguous 144-float conv-layout runs.
// ---------------------------------------------------------------------------
__global__ __launch_bounds__(256) void gemm2_kernel(
    const __bf16* __restrict__ H, const __bf16* __restrict__ W2T,
    const float* __restrict__ b2, const int* __restrict__ row_base,
    const int* __restrict__ ci9, float* __restrict__ out)
{
    // Single shared block => LDS offset 0; byte offsets below match the
    // descriptors passed to the TDM.
    __shared__ __align__(16) char lds_raw[64 * 1024];
    // [0,8K)=B0 [8K,16K)=B1 [16K,24K)=A0 [24K,32K)=A1 ; epilogue aliases all.
    __bf16* ldsB[2] = { (__bf16*)(lds_raw),         (__bf16*)(lds_raw + 8192)  };
    __bf16* ldsA[2] = { (__bf16*)(lds_raw + 16384), (__bf16*)(lds_raw + 24576) };
    const unsigned offB[2] = { 0u, 8192u };
    const unsigned offA[2] = { 16384u, 24576u };

    const int lane = threadIdx.x & 31;
    const int wave = threadIdx.x >> 5;
    const int hi   = lane >> 4;
    const int ln16 = lane & 15;
    const int n0   = blockIdx.x * 128;     // N base of WG tile
    const int mwg  = blockIdx.y * 128;     // M base of WG tile
    const int m0   = mwg + wave * 16;      // wave's 16 rows

    v8f acc[8] = {};

    // Prologue: DMA chunk 0 (TENSORcnt = 2)
    if (wave == 0) {
        tdm_load_2d_bf16(offA[0], H   + (size_t)mwg * HID, HID, MTOT, 32, 128, HID);
        tdm_load_2d_bf16(offB[0], W2T + (size_t)n0  * HID, HID, OUTD, 32, 128, HID);
    }

    int cur = 0;
    for (int kc = 0; kc < HID / 32; ++kc) {
        __syncthreads();                       // buf[cur^1] reads from kc-1 done
        if (wave == 0) {
            if (kc + 1 < HID / 32) {
                tdm_load_2d_bf16(offA[cur ^ 1],
                    H   + (size_t)mwg * HID + (kc + 1) * 32, HID, MTOT, 32, 128, HID);
                tdm_load_2d_bf16(offB[cur ^ 1],
                    W2T + (size_t)n0  * HID + (kc + 1) * 32, HID, OUTD, 32, 128, HID);
                __builtin_amdgcn_s_wait_tensorcnt(2);   // chunk kc landed
            } else {
                __builtin_amdgcn_s_wait_tensorcnt(0);   // last chunk landed
            }
        }
        __syncthreads();                       // publish chunk kc to all waves

        const __bf16* Ab = ldsA[cur];
        const __bf16* Bb = ldsB[cur];
        v8bf alo = *(const v8bf*)(Ab + (wave * 16 + ln16) * 32 + hi * 16);
        v8bf ahi = *(const v8bf*)(Ab + (wave * 16 + ln16) * 32 + 8 + hi * 16);
        v16bf a = __builtin_shufflevector(alo, ahi,
            0,1,2,3,4,5,6,7,8,9,10,11,12,13,14,15);
#pragma unroll
        for (int nt = 0; nt < 8; ++nt) {
            v16bf b = *(const v16bf*)(Bb + (nt * 16 + ln16) * 32 + hi * 16);
            acc[nt] = __builtin_amdgcn_wmma_f32_16x16x32_bf16(
                false, a, false, b, (short)0, acc[nt], false, false);
        }
        cur ^= 1;
    }

    __syncthreads();                           // loop reads done before aliasing
    float* smem = (float*)lds_raw;
    float* sw = smem + wave * (16 * 128);
#pragma unroll
    for (int nt = 0; nt < 8; ++nt)
#pragma unroll
        for (int r = 0; r < 8; ++r)
            sw[(hi * 8 + r) * 128 + nt * 16 + ln16] = acc[nt][r];
    __syncthreads();

    float bb[4];
#pragma unroll
    for (int c = 0; c < 4; ++c) bb[c] = b2[n0 + lane * 4 + c];

    for (int row = 0; row < 16; ++row) {
        int m  = m0 + row;
        int rb = row_base[m];
        int c9 = ci9[m];
#pragma unroll
        for (int c = 0; c < 4; ++c) {
            int n = n0 + lane * 4 + c;
            float v = sw[row * 128 + lane * 4 + c] + bb[c];
            int i = n / 144;          // which c_out sub-row (0..15)
            int r = n - i * 144;      // offset within the 144-float run
            out[(size_t)rb + (size_t)i * c9 + r] = v;
        }
    }
}

// ---------------------------------------------------------------------------
// Host-side orchestration
// ---------------------------------------------------------------------------
extern "C" void kernel_launch(void* const* d_in, const int* in_sizes, int n_in,
                              void* d_out, int out_size, void* d_ws, size_t ws_size,
                              hipStream_t stream) {
    (void)in_sizes; (void)out_size; (void)ws_size;

    const float* ie = (const float*)d_in[0];
    const float* W1 = (const float*)d_in[1];
    const float* b1 = (const float*)d_in[2];
    const float* W2 = (const float*)d_in[3];
    const float* b2 = (const float*)d_in[4];
    float* out = (float*)d_out;

    // Workspace carve-out (256B aligned): ~32 MB total
    char* ws = (char*)d_ws;
    size_t off = 0;
    auto carve = [&](size_t bytes) -> void* {
        void* p = ws + off;
        off = (off + bytes + 255) & ~(size_t)255;
        return p;
    };
    __bf16* W2T     = (__bf16*)carve((size_t)OUTD * HID * 2);   // 4.72 MB
    __bf16* W1T     = (__bf16*)carve((size_t)HID * 64 * 2);     // 128 KB
    __bf16* Pbf     = (__bf16*)carve((size_t)MTOT * 64 * 2);    // 1.56 MB
    float*  hbias   = (float*) carve((size_t)HID * 4);
    int*    rowbase = (int*)   carve((size_t)MTOT * 4);
    int*    ci9     = (int*)   carve((size_t)MTOT * 4);
    __bf16* Hbf     = (__bf16*)carve((size_t)MTOT * HID * 2);   // 24.9 MB

    // --- prep ---
    w2t_kernel<<<(OUTD * HID + 255) / 256, 256, 0, stream>>>(W2, W2T);
    w1t_kernel<<<(HID * 64 + 255) / 256, 256, 0, stream>>>(W1, W1T);
    hbias_kernel<<<(HID + 255) / 256, 256, 0, stream>>>(ie, W1, b1, hbias);
    meta_kernel<<<(MTOT + 255) / 256, 256, 0, stream>>>(rowbase, ci9);

    if (n_in >= 5 + NLAYER) {
        // pos_embs passed as 13 separate inputs
        static const int rows[NLAYER] = {64, 128, 256, 512, 1024, 2048, 4096,
                                         2048, 1024, 512, 256, 128, 64};
        int start = 0;
        for (int L = 0; L < NLAYER; ++L) {
            int cnt = rows[L] * 64;
            cvt_kernel<<<(cnt + 255) / 256, 256, 0, stream>>>(
                (const float*)d_in[5 + L], Pbf + (size_t)start * 64, cnt);
            start += rows[L];
        }
    } else {
        // tuple flattened into one contiguous input (same layout)
        cvt_kernel<<<(MTOT * 64 + 255) / 256, 256, 0, stream>>>(
            (const float*)d_in[5], Pbf, MTOT * 64);
    }

    // --- GEMM1: pos contribution + hbias + relu -> H (bf16) ---
    gemm1_kernel<<<dim3(HID / 128, MTOT / 128), 256, 0, stream>>>(
        Pbf, W1T, hbias, Hbf);

    // --- GEMM2: H @ W2 + b2, scattered into conv-weight layout ---
    gemm2_kernel<<<dim3(OUTD / 128, MTOT / 128), 256, 0, stream>>>(
        Hbf, W2T, b2, rowbase, ci9, out);
}
